// _LSTM_14551349199285
// MI455X (gfx1250) — matrix operand order repaired and checked by reading, not compiled, not run on hardware
//
#include <hip/hip_runtime.h>
#include <hip/hip_bf16.h>
#include <stdint.h>

constexpr int kB  = 64;     // batch
constexpr int kS  = 512;    // seq len
constexpr int kIN = 1024;   // input dim
constexpr int kHZ = 1024;   // hidden dim
constexpr int kG  = 4096;   // 4*HZ (gates)

typedef __attribute__((ext_vector_type(16))) __bf16 v16bf;
typedef __attribute__((ext_vector_type(8)))  __bf16 v8bf;
typedef __attribute__((ext_vector_type(8)))  float  v8f;

// ---------------------------------------------------------------------------
// CDNA5 async global->LDS copy (ASYNCcnt-tracked, ISA cdna5 ch.10 / 15.18.3)
// ---------------------------------------------------------------------------
__device__ __forceinline__ void async_copy_b128(const __bf16* gsrc,
                                                const __bf16* lds_dst) {
  // VDST = 32-bit LDS byte address (wave-relative), VADDR = 64-bit global.
  uint32_t loff = (uint32_t)(size_t)(const void*)lds_dst;
  asm volatile("global_load_async_to_lds_b128 %0, %1, off"
               :
               : "v"(loff), "v"(gsrc)
               : "memory");
}

__device__ __forceinline__ void wait_async_le4() {
#if __has_builtin(__builtin_amdgcn_s_wait_asynccnt)
  __builtin_amdgcn_s_wait_asynccnt(4);
#else
  asm volatile("s_wait_asynccnt 0x4" ::: "memory");
#endif
}
__device__ __forceinline__ void wait_async_0() {
#if __has_builtin(__builtin_amdgcn_s_wait_asynccnt)
  __builtin_amdgcn_s_wait_asynccnt(0);
#else
  asm volatile("s_wait_asynccnt 0x0" ::: "memory");
#endif
}

// ---------------------------------------------------------------------------
// WMMA fragment helpers (wave32, v_wmma_f32_16x16x32_bf16)
// A 16x32 bf16: lane L = row L%16; elem e at k=16*(L/16)+e.
// B 32x16: lane L = col L%16; elem e at k=16*(L/16)+e.
// C = A * W^T with W row-major [N][K] -> B frag = 32 contiguous bytes of row n.
// ---------------------------------------------------------------------------
__device__ __forceinline__ v16bf make_a_frag(const __bf16* p16, const __bf16* p32) {
  v8bf lo = *(const v8bf*)p16;
  v8bf hi = *(const v8bf*)p32;
  return __builtin_shufflevector(lo, hi, 0, 1, 2, 3, 4, 5, 6, 7,
                                 8, 9, 10, 11, 12, 13, 14, 15);
}

__device__ __forceinline__ v16bf load_a_frag_g(const __bf16* __restrict__ a_kk,
                                               int lda, int lane) {
  const int m = lane & 15, g = lane >> 4;
  const __bf16* p = a_kk + (size_t)m * lda + (g << 4);
  return make_a_frag(p, p + 8);
}
__device__ __forceinline__ v16bf load_b_frag_g(const __bf16* __restrict__ w_kk,
                                               int ldw, int lane) {
  const int n = lane & 15, g = lane >> 4;
  return *(const v16bf*)(w_kk + (size_t)n * ldw + (g << 4));
}

constexpr int kLdsStride = 40;  // 32 data + 8 pad bf16 -> conflict-free rows

__device__ __forceinline__ v16bf lds_a_frag(const __bf16* base, int lane) {
  const int m = lane & 15, g = lane >> 4;
  const __bf16* p = base + m * kLdsStride + (g << 4);
  return make_a_frag(p, p + 8);
}
__device__ __forceinline__ v16bf lds_b_frag(const __bf16* base, int lane) {
  const int n = lane & 15, g = lane >> 4;
  return *(const v16bf*)(base + n * kLdsStride + (g << 4));
}

__device__ __forceinline__ v8f wmma_bf16(v16bf a, v16bf b, v8f c) {
  return __builtin_amdgcn_wmma_f32_16x16x32_bf16(
      false, a, false, b, (short)0, c, false, false);
}

__device__ __forceinline__ float fast_sigmoid(float x) {
  return 1.0f / (1.0f + __expf(-x));
}
__device__ __forceinline__ float fast_tanh(float x) {
  return 1.0f - 2.0f / (__expf(2.0f * x) + 1.0f);
}

// ---------------------------------------------------------------------------
// Precision conversion / layout kernels
// ---------------------------------------------------------------------------
__global__ void convert_weights_kernel(const float* __restrict__ wi,
                                       const float* __restrict__ wh,
                                       __bf16* __restrict__ wi_bf,
                                       __bf16* __restrict__ wh_bf) {
  size_t i = (size_t)blockIdx.x * blockDim.x + threadIdx.x;
  const size_t n = (size_t)kG * kIN;
  if (i < n)
    wi_bf[i] = (__bf16)wi[i];
  else if (i < 2 * n)
    wh_bf[i - n] = (__bf16)wh[i - n];
}

// x [B][S][IN] fp32 -> x_bf [S][B][IN] bf16 (row r = s*B + b)
__global__ void convert_x_kernel(const float* __restrict__ x,
                                 __bf16* __restrict__ x_bf) {
  size_t i = (size_t)blockIdx.x * blockDim.x + threadIdx.x;
  if (i >= (size_t)kS * kB * kIN) return;
  int k = (int)(i % kIN);
  size_t sb = i / kIN;
  int b = (int)(sb % kB);
  int s = (int)(sb / kB);
  x_bf[i] = (__bf16)x[((size_t)b * kS + s) * kIN + k];
}

__global__ void init_state_kernel(const float* __restrict__ h0,
                                  const float* __restrict__ c0,
                                  __bf16* __restrict__ hbuf,
                                  float* __restrict__ c_ws) {
  int i = blockIdx.x * blockDim.x + threadIdx.x;
  if (i < kB * kHZ) {
    hbuf[i] = (__bf16)h0[i];
    c_ws[i] = c0[i];
  }
}

// ---------------------------------------------------------------------------
// Phase 1: x_ih = x_bf @ i2h_w^T + b.  M=32768, N=4096, K=1024.
// Block (256 thr, 8 waves) computes 128x128; per 32-K chunk it async-stages
// A(128x32) and B(128x32) into double-buffered LDS; each wave computes a
// 32x64 register tile (8 WMMAs/chunk, B frags reused across 2 A frags).
// Explicit ping/pong body with named accumulators so the allocator keeps
// each accumulator pinned (no cross-phase v_mov shuffles / hazard NOPs).
// ---------------------------------------------------------------------------
constexpr int kBlkM = 128, kBlkN = 128, kChunk = 32;
constexpr int kNChunks = kIN / kChunk;  // 32

__device__ __forceinline__ void stage_chunk(const __bf16* __restrict__ ga,
                                            const __bf16* __restrict__ gb,
                                            __bf16* la, __bf16* lb,
                                            int kk, int tid) {
  // 128 rows x 32 k per array; 4 x b128 segments per row; 512+512 transfers.
#pragma unroll
  for (int h = 0; h < 2; ++h) {
    const int j = tid + h * 256;       // 0..511
    const int row = j >> 2, seg = j & 3;
    async_copy_b128(ga + (size_t)row * kIN + kk + seg * 8,
                    la + row * kLdsStride + seg * 8);
    async_copy_b128(gb + (size_t)row * kIN + kk + seg * 8,
                    lb + row * kLdsStride + seg * 8);
  }
}

__global__ __launch_bounds__(256) void gemm_xih_kernel(
    const __bf16* __restrict__ x_bf,   // [S*B][IN]
    const __bf16* __restrict__ wi_bf,  // [4H][IN]
    const float* __restrict__ i2h_b,   // [4H]
    float* __restrict__ x_ih) {        // [S*B][4H]
  __shared__ __bf16 la[2][kBlkM * kLdsStride];  // 2 x 10 KB
  __shared__ __bf16 lb[2][kBlkN * kLdsStride];  // 2 x 10 KB

  const int tid  = threadIdx.x;
  const int lane = tid & 31;
  const int wave = tid >> 5;   // 0..7
  const int wm   = wave >> 1;  // 0..3 -> local rows wm*32
  const int wn   = wave & 1;   // 0..1 -> local cols wn*64
  const int bM   = blockIdx.x >> 5;  // 0..255
  const int bN   = blockIdx.x & 31;  // 0..31
  const int row0 = bM * kBlkM;
  const int col0 = bN * kBlkN;

  const __bf16* ga = x_bf + (size_t)row0 * kIN;
  const __bf16* gb = wi_bf + (size_t)col0 * kIN;

  v8f c00{}, c01{}, c02{}, c03{};  // A-subtile 0 x B-subtiles 0..3
  v8f c10{}, c11{}, c12{}, c13{};  // A-subtile 1 x B-subtiles 0..3

#define COMPUTE_CHUNK(LA, LB)                                          \
  {                                                                    \
    const __bf16* abase = (LA) + (wm * 32) * kLdsStride;               \
    const __bf16* bbase = (LB) + (wn * 64) * kLdsStride;               \
    v16bf a0 = lds_a_frag(abase, lane);                                \
    v16bf a1 = lds_a_frag(abase + 16 * kLdsStride, lane);              \
    v16bf b0 = lds_b_frag(bbase, lane);                                \
    v16bf b1 = lds_b_frag(bbase + 16 * kLdsStride, lane);              \
    v16bf b2 = lds_b_frag(bbase + 32 * kLdsStride, lane);              \
    v16bf b3 = lds_b_frag(bbase + 48 * kLdsStride, lane);              \
    c00 = wmma_bf16(a0, b0, c00);                                      \
    c10 = wmma_bf16(a1, b0, c10);                                      \
    c01 = wmma_bf16(a0, b1, c01);                                      \
    c11 = wmma_bf16(a1, b1, c11);                                      \
    c02 = wmma_bf16(a0, b2, c02);                                      \
    c12 = wmma_bf16(a1, b2, c12);                                      \
    c03 = wmma_bf16(a0, b3, c03);                                      \
    c13 = wmma_bf16(a1, b3, c13);                                      \
  }

  stage_chunk(ga, gb, la[0], lb[0], 0, tid);

#pragma clang loop unroll(disable)
  for (int c = 0; c < kNChunks; c += 2) {
    // ---- phase A: compute chunk c from buf0, stage chunk c+1 into buf1 ----
    stage_chunk(ga, gb, la[1], lb[1], (c + 1) * kChunk, tid);
    wait_async_le4();  // 4 newer in flight -> chunk c (older) has landed
    __syncthreads();
    COMPUTE_CHUNK(la[0], lb[0]);
    __syncthreads();   // buf0 reads done everywhere before re-staging it
    // ---- phase B: compute chunk c+1 from buf1, stage chunk c+2 into buf0 --
    if (c + 2 < kNChunks) {
      stage_chunk(ga, gb, la[0], lb[0], (c + 2) * kChunk, tid);
      wait_async_le4();
    } else {
      wait_async_0();
    }
    __syncthreads();
    COMPUTE_CHUNK(la[1], lb[1]);
    __syncthreads();
  }
#undef COMPUTE_CHUNK

  const int hf = lane >> 4, nn = lane & 15;
  const v8f* accs[2][4] = {{&c00, &c01, &c02, &c03}, {&c10, &c11, &c12, &c13}};
#pragma unroll
  for (int s = 0; s < 2; ++s) {
#pragma unroll
    for (int q = 0; q < 4; ++q) {
      const int cc = col0 + wn * 64 + q * 16 + nn;
      const float bias = i2h_b[cc];
      const v8f av = *accs[s][q];
#pragma unroll
      for (int r = 0; r < 8; ++r) {
        const int rr = row0 + wm * 32 + s * 16 + r + (hf << 3);
        x_ih[(size_t)rr * kG + cc] = av[r] + bias;
      }
    }
  }
}

// ---------------------------------------------------------------------------
// Phase 2: one fused kernel per timestep. Each wave computes a 16(batch) x
// 16(hidden) tile for ALL FOUR gates, so the i/f/o/g quartet per (b,j) is
// wave-local -> gates + c/h update fused, no preact round-trip.
// ---------------------------------------------------------------------------
__global__ __launch_bounds__(128) void lstm_step_kernel(
    const __bf16* __restrict__ h_prev,  // [B][HZ] bf16 (ping)
    __bf16* __restrict__ h_next,        // [B][HZ] bf16 (pong)
    const __bf16* __restrict__ wh_bf,   // [4H][HZ]
    const float* __restrict__ x_ih_t,   // [B][4H] at timestep t
    const float* __restrict__ h2h_b,    // [4H]
    float* __restrict__ c_ws,           // [B][HZ] fp32, in/out
    float* __restrict__ out_seq,        // [B][S][HZ]
    float* __restrict__ out_ht,         // [B][HZ]
    float* __restrict__ out_ct,         // [B][HZ]
    int t) {
  const int lane  = threadIdx.x & 31;
  const int wave  = blockIdx.x * 4 + (threadIdx.x >> 5);  // 0..255
  const int tileJ = wave & 63;
  const int tileM = wave >> 6;  // 0..3
  const int row0  = tileM << 4;
  const int j0    = tileJ << 4;

  const __bf16* a_ptr = h_prev + (size_t)row0 * kHZ;
  const __bf16* w_ptr = wh_bf + (size_t)j0 * kHZ;
  constexpr size_t kGateStride = (size_t)kHZ * kHZ;

  v8f acc[4] = {};  // 0:i 1:f 2:o 3:g
#pragma unroll 2
  for (int kk = 0; kk < kHZ; kk += 32) {
    v16bf a = load_a_frag_g(a_ptr + kk, kHZ, lane);
#pragma unroll
    for (int q = 0; q < 4; ++q) {
      v16bf bm = load_b_frag_g(w_ptr + (size_t)q * kGateStride + kk, kHZ, lane);
      acc[q] = wmma_bf16(a, bm, acc[q]);
    }
  }

  const int hf = lane >> 4, nn = lane & 15;
  const int j = j0 + nn;
#pragma unroll
  for (int r = 0; r < 8; ++r) {
    const int b = row0 + r + (hf << 3);
    const size_t xo = (size_t)b * kG + j;
    float pi = acc[0][r] + x_ih_t[xo]           + h2h_b[j];
    float pf = acc[1][r] + x_ih_t[xo + kHZ]     + h2h_b[j + kHZ];
    float po = acc[2][r] + x_ih_t[xo + 2 * kHZ] + h2h_b[j + 2 * kHZ];
    float pg = acc[3][r] + x_ih_t[xo + 3 * kHZ] + h2h_b[j + 3 * kHZ];
    // faithful to reference: sigmoid applied twice to i/f/o pre-activations
    float it = fast_sigmoid(fast_sigmoid(pi));
    float ft = fast_sigmoid(fast_sigmoid(pf));
    float ot = fast_sigmoid(fast_sigmoid(po));
    float gt = fast_tanh(pg);
    const size_t ci = (size_t)b * kHZ + j;
    float cn = ft * c_ws[ci] + it * gt;
    c_ws[ci] = cn;
    float h = ot * fast_tanh(cn);
    out_seq[((size_t)b * kS + t) * kHZ + j] = h;
    h_next[ci] = (__bf16)h;
    if (t == kS - 1) {
      out_ht[ci] = h;
      out_ct[ci] = cn;
    }
  }
}

// ---------------------------------------------------------------------------
extern "C" void kernel_launch(void* const* d_in, const int* in_sizes, int n_in,
                              void* d_out, int out_size, void* d_ws,
                              size_t ws_size, hipStream_t stream) {
  const float* x     = (const float*)d_in[0];
  const float* h0    = (const float*)d_in[1];
  const float* c0    = (const float*)d_in[2];
  const float* i2h_w = (const float*)d_in[3];
  const float* i2h_b = (const float*)d_in[4];
  const float* h2h_w = (const float*)d_in[5];
  const float* h2h_b = (const float*)d_in[6];
  float* out = (float*)d_out;

  // workspace layout
  char* ws = (char*)d_ws;
  __bf16* x_bf  = (__bf16*)ws;  ws += (size_t)kS * kB * kIN * 2;  //  64 MB
  __bf16* wi_bf = (__bf16*)ws;  ws += (size_t)kG * kIN * 2;       //   8 MB
  __bf16* wh_bf = (__bf16*)ws;  ws += (size_t)kG * kHZ * 2;       //   8 MB
  __bf16* hbuf0 = (__bf16*)ws;  ws += (size_t)kB * kHZ * 2;
  __bf16* hbuf1 = (__bf16*)ws;  ws += (size_t)kB * kHZ * 2;
  float*  c_ws  = (float*)ws;   ws += (size_t)kB * kHZ * 4;
  float*  x_ih  = (float*)ws;   // [S][B][4H] fp32 = 512 MB

  {
    size_t n = 2ull * kG * kIN;
    convert_weights_kernel<<<(unsigned)((n + 255) / 256), 256, 0, stream>>>(
        i2h_w, h2h_w, wi_bf, wh_bf);
  }
  {
    size_t n = (size_t)kS * kB * kIN;
    convert_x_kernel<<<(unsigned)((n + 255) / 256), 256, 0, stream>>>(x, x_bf);
  }
  init_state_kernel<<<(kB * kHZ + 255) / 256, 256, 0, stream>>>(h0, c0, hbuf0,
                                                                c_ws);

  // Phase 1: grid = (32768/128) x (4096/128) = 256 x 32 = 8192 blocks
  gemm_xih_kernel<<<8192, 256, 0, stream>>>(x_bf, wi_bf, i2h_b, x_ih);

  // Phase 2: sequential scan, one fused kernel per step (graph-friendly)
  float* out_ht = out + (size_t)kB * kS * kHZ;
  float* out_ct = out_ht + (size_t)kB * kHZ;
  for (int t = 0; t < kS; ++t) {
    const __bf16* hp = (t & 1) ? hbuf1 : hbuf0;
    __bf16* hn       = (t & 1) ? hbuf0 : hbuf1;
    lstm_step_kernel<<<64, 128, 0, stream>>>(
        hp, hn, wh_bf, x_ih + (size_t)t * kB * kG, h2h_b, c_ws, out, out_ht,
        out_ct, t);
  }
}
